// GNNPredictor_34024730919215
// MI455X (gfx1250) — matrix operand order repaired and checked by reading, not compiled, not run on hardware
//
#include <hip/hip_runtime.h>
#include <hip/hip_bf16.h>

// ---------------- problem constants (from reference) ----------------
constexpr int kN   = 20000;   // nodes
constexpr int kE   = 320000;  // edges (before self loops)
constexpr int kET  = kE + kN; // edges + self loops
constexpr int kG   = 64;      // graphs
constexpr int kDIN = 256;
constexpr int kD   = 128;
constexpr int kH   = 4;
constexpr int kHD  = kH * kD; // 512
constexpr int kL   = 3;
constexpr float kEPS  = 1e-5f;
constexpr float kNEG  = 0.2f;

// ---------------- WMMA types ----------------
typedef __attribute__((ext_vector_type(16))) __bf16 v16bf;
typedef __attribute__((ext_vector_type(8)))  float  v8f;

// ---------------- helpers ----------------
__device__ __forceinline__ void atomicMaxF(float* addr, float val) {
  // sign-aware int punning; works with -inf init (0xFF800000)
  if (val >= 0.0f) atomicMax((int*)addr, __float_as_int(val));
  else             atomicMin((unsigned int*)addr, __float_as_uint(val));
}

__device__ __forceinline__ void edge_sd(const int* __restrict__ ei, int e, int& s, int& t) {
  if (e < kE) { s = ei[e]; t = ei[kE + e]; }
  else        { s = e - kE; t = e - kE; }   // self loop
}

// ---------------- generic fill ----------------
__global__ void fill_f32(float* __restrict__ p, float v, size_t n) {
  size_t i = (size_t)blockIdx.x * blockDim.x + threadIdx.x;
  size_t stride = (size_t)gridDim.x * blockDim.x;
  for (; i < n; i += stride) p[i] = v;
}

// ================= WMMA GEMM with LDS-staged, fragment-swizzled weights =================
// C[M,Nout] = act(A[M,K] @ W[Nout,K]^T + bias)
//
// Block = (32 lanes, 8 waves). Each block owns NTB n-tiles (16 cols each) and
// grid-strides over m-tiles. The weight panel (NTB*16 x K) is converted f32->bf16
// ONCE per block into LDS, already laid out in B-fragment order, so each WMMA's
// B operand is a single 32-byte LDS read per lane (2x ds_load_b128).
//
// Fragment layouts used here (wave32; A/B use contiguous k, not the ISA table order):
//   A (16x32 bf16): lane (g=l>>4, r=l&15) holds row r; elem i -> K=k0+16g+i
//   B (32x16 bf16): lane holds col r;      elem i -> K=k0+16g+i
//   C/D (16x16 f32): lane col r; vgpr v -> row v+8g
template <int K, int NTB>
__global__ void wmma_gemm_lds(const float* __restrict__ A, const float* __restrict__ W,
                              const float* __restrict__ bias, float* __restrict__ C,
                              int M, int Nout, int do_relu) {
  constexpr int K32   = K / 32;                 // k-blocks
  constexpr int FRAGS = NTB * K32;              // B fragments held in LDS
  __shared__ __attribute__((aligned(32))) __bf16 smem[FRAGS * 32 * 16];  // 64 KB

  const int lane = threadIdx.x;
  const int g = lane >> 4;
  const int r = lane & 15;
  const int tid = threadIdx.y * 32 + lane;      // 0..255
  const int ntile0 = blockIdx.y * NTB;          // first n-tile of this block

  // ---- cooperative LDS fill: weights in B-fragment order, bf16 ----
  // slot = fragment * 32 + lane_in_fragment; each slot is 16 contiguous bf16.
  for (int slot = tid; slot < FRAGS * 32; slot += 256) {
    const int f  = slot >> 5;
    const int sl = slot & 31;
    const int nt = f / K32;
    const int kb = f - nt * K32;
    const int n  = (ntile0 + nt) * 16 + (sl & 15);
    const int k0 = kb * 32 + 16 * (sl >> 4);
    const float* __restrict__ wp = W + (size_t)n * K + k0;
    v16bf tmp;
#pragma unroll
    for (int i = 0; i < 16; ++i) tmp[i] = (__bf16)wp[i];
    *(v16bf*)&smem[(size_t)slot * 16] = tmp;
  }
  __syncthreads();

  const int mtiles = M / 16;
  for (int mt = blockIdx.x; mt < mtiles; mt += gridDim.x) {
    // ---- load & convert all A fragments for this m-tile once ----
    const float* __restrict__ arow = A + (size_t)(mt * 16 + r) * K;
    v16bf afr[K32];
#pragma unroll
    for (int kb = 0; kb < K32; ++kb) {
      const int k0 = kb * 32;
#pragma unroll
      for (int i = 0; i < 16; ++i) {
        afr[kb][i]     = (__bf16)arow[k0 + 16 * g + i];
      }
    }
    // ---- each wave sweeps its n-tiles, B straight from LDS ----
    for (int nt = threadIdx.y; nt < NTB; nt += 8) {
      v8f acc = {};
#pragma unroll
      for (int kb = 0; kb < K32; ++kb) {
        const v16bf b = *(const v16bf*)&smem[((size_t)(nt * K32 + kb) * 32 + lane) * 16];
        acc = __builtin_amdgcn_wmma_f32_16x16x32_bf16(
            /*neg_a=*/false, afr[kb], /*neg_b=*/false, b,
            /*c_mod=*/(short)0, acc, /*reuse_a=*/false, /*reuse_b=*/false);
      }
      const int n0 = (ntile0 + nt) * 16;
      const float bv = bias ? bias[n0 + r] : 0.0f;
#pragma unroll
      for (int v = 0; v < 8; ++v) {
        const int m = mt * 16 + v + 8 * g;
        float val = acc[v] + bv;
        if (do_relu) val = fmaxf(val, 0.0f);
        C[(size_t)m * Nout + (n0 + r)] = val;
      }
    }
  }
}

// ---------------- per-head attention dots: a_src/a_dst [N,H] ----------------
__global__ void att_dots(const float* __restrict__ hproj,
                         const float* __restrict__ att_src, const float* __restrict__ att_dst,
                         float* __restrict__ a_src, float* __restrict__ a_dst) {
  int idx = blockIdx.x * blockDim.x + threadIdx.x;
  if (idx >= kN * kH) return;
  const int n = idx / kH, hh = idx % kH;
  const float* __restrict__ hp = hproj + (size_t)n * kHD + hh * kD;
  const float* __restrict__ ws = att_src + hh * kD;
  const float* __restrict__ wd = att_dst + hh * kD;
  float as = 0.0f, ad = 0.0f;
#pragma unroll 4
  for (int d = 0; d < kD; ++d) { float v = hp[d]; as += v * ws[d]; ad += v * wd[d]; }
  a_src[idx] = as;
  a_dst[idx] = ad;
}

// ---------------- edge logits -> segment max over dst ----------------
__global__ void edge_max(const int* __restrict__ ei,
                         const float* __restrict__ a_src, const float* __restrict__ a_dst,
                         float* __restrict__ emax) {
  int e = blockIdx.x * blockDim.x + threadIdx.x;
  if (e >= kET) return;
  int s, t; edge_sd(ei, e, s, t);
#pragma unroll
  for (int hh = 0; hh < kH; ++hh) {
    float v = a_src[s * kH + hh] + a_dst[t * kH + hh];
    v = v > 0.0f ? v : kNEG * v;
    atomicMaxF(&emax[t * kH + hh], v);
  }
}

// ---------------- edge weights: w = exp(e - max), denom = segment_sum ----------------
__global__ void edge_weights(const int* __restrict__ ei,
                             const float* __restrict__ a_src, const float* __restrict__ a_dst,
                             const float* __restrict__ emax,
                             float* __restrict__ wbuf, float* __restrict__ denom) {
  int e = blockIdx.x * blockDim.x + threadIdx.x;
  if (e >= kET) return;
  int s, t; edge_sd(ei, e, s, t);
#pragma unroll
  for (int hh = 0; hh < kH; ++hh) {
    float v = a_src[s * kH + hh] + a_dst[t * kH + hh];
    v = v > 0.0f ? v : kNEG * v;
    float w = __expf(v - emax[t * kH + hh]);
    wbuf[(size_t)e * kH + hh] = w;
    atomicAdd(&denom[t * kH + hh], w);
  }
}

// ---------------- message scatter: aggr[t,h,:] += h[s,h,:] * alpha ----------------
// one block (128 lanes) per edge; hproj (41MB) lives in the 192MB L2.
__global__ void edge_scatter(const int* __restrict__ ei,
                             const float* __restrict__ hproj,
                             const float* __restrict__ wbuf, const float* __restrict__ denom,
                             float* __restrict__ aggr) {
  const int e = blockIdx.x;
  const int d = threadIdx.x;
  int s, t; edge_sd(ei, e, s, t);
  // pull the gathered source row toward the WGP early (global_prefetch_b8)
  __builtin_prefetch(hproj + (size_t)s * kHD + d, 0, 1);
#pragma unroll
  for (int hh = 0; hh < kH; ++hh) {
    const float alpha = wbuf[(size_t)e * kH + hh] / (denom[t * kH + hh] + 1e-16f);
    atomicAdd(&aggr[((size_t)t * kH + hh) * kD + d],
              hproj[(size_t)s * kHD + hh * kD + d] * alpha);
  }
}

// ---------------- head-mean + bias, accumulate BN stats ----------------
__global__ void node_finalize(const float* __restrict__ aggr, const float* __restrict__ gat_b,
                              float* __restrict__ h, float* __restrict__ bnstat,
                              int rows_per_block) {
  const int d = threadIdx.x;            // 0..127, channel per lane
  const int r0 = blockIdx.x * rows_per_block;
  float s = 0.0f, s2 = 0.0f;
  for (int i = 0; i < rows_per_block; ++i) {
    const int n = r0 + i;
    if (n >= kN) break;
    float v = 0.0f;
#pragma unroll
    for (int hh = 0; hh < kH; ++hh) v += aggr[((size_t)n * kH + hh) * kD + d];
    v = v * 0.25f + gat_b[d];
    h[(size_t)n * kD + d] = v;
    s += v; s2 += v * v;
  }
  atomicAdd(&bnstat[d], s);
  atomicAdd(&bnstat[kD + d], s2);
}

// ---------------- BN (biased var) + ReLU, in place ----------------
__global__ void bn_apply(float* __restrict__ h, const float* __restrict__ bnstat,
                         const float* __restrict__ g, const float* __restrict__ b) {
  int idx = blockIdx.x * blockDim.x + threadIdx.x;
  if (idx >= kN * kD) return;
  const int d = idx & (kD - 1);
  const float mu  = bnstat[d] * (1.0f / kN);
  const float var = bnstat[kD + d] * (1.0f / kN) - mu * mu;
  float v = (h[idx] - mu) * rsqrtf(var + kEPS) * g[d] + b[d];
  h[idx] = fmaxf(v, 0.0f);
}

// ---------------- global mean pool (batch is sorted -> local accumulation) ----------------
__global__ void pool_kernel(const float* __restrict__ h, const int* __restrict__ batch,
                            float* __restrict__ pooled, float* __restrict__ cnt,
                            int rows_per_block) {
  const int d = threadIdx.x;            // 0..127
  const int r0 = blockIdx.x * rows_per_block;
  int cur = -1;
  float acc = 0.0f;
  float c = 0.0f;
  for (int i = 0; i < rows_per_block; ++i) {
    const int n = r0 + i;
    if (n >= kN) break;
    const int gid = batch[n];
    if (gid != cur) {
      if (cur >= 0) {
        atomicAdd(&pooled[cur * kD + d], acc);
        if (d == 0) atomicAdd(&cnt[cur], c);
      }
      cur = gid; acc = 0.0f; c = 0.0f;
    }
    acc += h[(size_t)n * kD + d];
    c += 1.0f;
  }
  if (cur >= 0) {
    atomicAdd(&pooled[cur * kD + d], acc);
    if (d == 0) atomicAdd(&cnt[cur], c);
  }
}

// ---------------- three MLP heads -> out[G,3] ----------------
__global__ void heads_kernel(const float* __restrict__ pooled, const float* __restrict__ cnt,
                             const float* __restrict__ hw1, const float* __restrict__ hb1,
                             const float* __restrict__ hw2, const float* __restrict__ hb2,
                             float* __restrict__ out) {
  int idx = blockIdx.x * blockDim.x + threadIdx.x;
  if (idx >= kG * 3) return;
  const int gidx = idx / 3, k = idx % 3;
  const float inv = 1.0f / fmaxf(cnt[gidx], 1.0f);
  const float* __restrict__ pg = pooled + gidx * kD;
  float o = hb2[k];
  for (int oo = 0; oo < kD / 2; ++oo) {
    float acc = hb1[k * (kD / 2) + oo];
    const float* __restrict__ w = hw1 + ((size_t)k * (kD / 2) + oo) * kD;
#pragma unroll 4
    for (int dd = 0; dd < kD; ++dd) acc += pg[dd] * inv * w[dd];
    o += fmaxf(acc, 0.0f) * hw2[k * (kD / 2) + oo];
  }
  out[gidx * 3 + k] = o;
}

// ---------------- host side ----------------
extern "C" void kernel_launch(void* const* d_in, const int* in_sizes, int n_in,
                              void* d_out, int out_size, void* d_ws, size_t ws_size,
                              hipStream_t stream) {
  const float* x       = (const float*)d_in[0];
  const int*   ei      = (const int*)  d_in[1];
  const int*   batch   = (const int*)  d_in[2];
  const float* ip_w    = (const float*)d_in[3];
  const float* ip_b    = (const float*)d_in[4];
  const float* lin_w   = (const float*)d_in[5];
  const float* att_src = (const float*)d_in[6];
  const float* att_dst = (const float*)d_in[7];
  const float* gat_b   = (const float*)d_in[8];
  const float* bn_g    = (const float*)d_in[9];
  const float* bn_b    = (const float*)d_in[10];
  const float* hw1     = (const float*)d_in[11];
  const float* hb1     = (const float*)d_in[12];
  const float* hw2     = (const float*)d_in[13];
  const float* hb2     = (const float*)d_in[14];

  // workspace carve-up (floats)
  float* ws     = (float*)d_ws;
  float* h      = ws;                              // N*D
  float* hproj  = h      + (size_t)kN * kD;        // N*HD
  float* aggr   = hproj  + (size_t)kN * kHD;       // N*HD
  float* a_src  = aggr   + (size_t)kN * kHD;       // N*H
  float* a_dst  = a_src  + (size_t)kN * kH;
  float* emax   = a_dst  + (size_t)kN * kH;
  float* denom  = emax   + (size_t)kN * kH;
  float* wbuf   = denom  + (size_t)kN * kH;        // ET*H
  float* bnstat = wbuf   + (size_t)kET * kH;       // 2*D
  float* pooled = bnstat + 2 * kD;                 // G*D
  float* cnt    = pooled + (size_t)kG * kD;        // G

  const dim3 gblock(32, 8);
  const float NEGINF = __builtin_huge_valf() * -1.0f;

  // input projection: h = relu(x @ ip_w^T + ip_b)   [20000,256]x[256,128]
  // one block covers all 8 n-tiles (weight panel 128x256 bf16 = 64KB LDS)
  wmma_gemm_lds<kDIN, 8><<<dim3(160, 1), gblock, 0, stream>>>(
      x, ip_w, ip_b, h, kN, kD, 1);

  for (int l = 0; l < kL; ++l) {
    // per-head projection: hproj = h @ lin_w[l]^T   [20000,128]x[128,512]
    // grid.y=2 halves of the 32 n-tiles (weight panel 256x128 bf16 = 64KB LDS each)
    wmma_gemm_lds<kD, 16><<<dim3(160, 2), gblock, 0, stream>>>(
        h, lin_w + (size_t)l * kHD * kD, nullptr, hproj, kN, kHD, 0);

    att_dots<<<(kN * kH + 255) / 256, 256, 0, stream>>>(
        hproj, att_src + (size_t)l * kH * kD, att_dst + (size_t)l * kH * kD, a_src, a_dst);

    fill_f32<<<256, 256, 0, stream>>>(emax, NEGINF, (size_t)kN * kH);
    fill_f32<<<256, 256, 0, stream>>>(denom, 0.0f, (size_t)kN * kH);

    edge_max<<<(kET + 255) / 256, 256, 0, stream>>>(ei, a_src, a_dst, emax);
    edge_weights<<<(kET + 255) / 256, 256, 0, stream>>>(ei, a_src, a_dst, emax, wbuf, denom);

    fill_f32<<<4096, 256, 0, stream>>>(aggr, 0.0f, (size_t)kN * kHD);
    edge_scatter<<<kET, kD, 0, stream>>>(ei, hproj, wbuf, denom, aggr);

    fill_f32<<<1, 256, 0, stream>>>(bnstat, 0.0f, (size_t)2 * kD);
    node_finalize<<<200, kD, 0, stream>>>(aggr, gat_b + (size_t)l * kD, h, bnstat, 100);
    bn_apply<<<(kN * kD + 255) / 256, 256, 0, stream>>>(
        h, bnstat, bn_g + (size_t)l * kD, bn_b + (size_t)l * kD);
  }

  fill_f32<<<64, 256, 0, stream>>>(pooled, 0.0f, (size_t)kG * kD + kG); // pooled + cnt contiguous
  pool_kernel<<<200, kD, 0, stream>>>(h, batch, pooled, cnt, 100);
  heads_kernel<<<1, 256, 0, stream>>>(pooled, cnt, hw1, hb1, hw2, hb2, (float*)d_out);
}
